// BiAttentionLayer_8735963480429
// MI455X (gfx1250) — matrix operand order repaired and checked by reading, not compiled, not run on hardware
//
#include <hip/hip_runtime.h>
#include <hip/hip_bf16.h>

// ---------------------------------------------------------------------------
// BiAttention (B=8, P=2048, Q=256, D=768), fp32 in/out, bf16 WMMA GEMMs.
// ---------------------------------------------------------------------------

typedef __attribute__((ext_vector_type(16))) __bf16 v16bf;
typedef __attribute__((ext_vector_type(4)))  __bf16 v4bf;
typedef __attribute__((ext_vector_type(8)))  float  v8f;
typedef __attribute__((ext_vector_type(4)))  unsigned int u32x4;
typedef __attribute__((ext_vector_type(8)))  int    i32x8;
typedef __attribute__((ext_vector_type(4)))  int    i32x4;

constexpr int B_ = 8;
constexpr int P_ = 2048;
constexpr int Q_ = 256;
constexpr int D_ = 768;
constexpr int OUTD = 4 * D_;            // 3072
constexpr float EPS_ = 1e-7f;
constexpr float NEGV = -10000000.0f;

constexpr int ROWS_PER_WAVE = 16;
constexpr int WAVES = 4;                // 128 threads/block
constexpr int ROWS_PER_WG = ROWS_PER_WAVE * WAVES;   // 64
constexpr int SIMLD = Q_ + 1;           // LDS row stride (pad)

// K index inside a 16-bit A-fragment (16x32): pair j (0..7), lane-half h.
// Contiguous order matching B loads: K = h*16 + 2j
__device__ __forceinline__ int a_kbase(int j, int h) {
  return h * 16 + 2 * j;
}

// ---------------------------------------------------------------------------
// Kernel 0: convert question fp32 -> bf16, row-major (qbf[b][q][d]) and
// transposed (qbfT[b][d][q]). Both feed contiguous 32B B-fragment loads.
// ---------------------------------------------------------------------------
__global__ __launch_bounds__(256) void
bi_convert_q_kernel(const float* __restrict__ quest,
                    __bf16* __restrict__ qbf,
                    __bf16* __restrict__ qbfT) {
  const size_t idx = (size_t)blockIdx.x * 256 + threadIdx.x;   // B*Q*D/4
  const size_t total = (size_t)B_ * Q_ * D_ / 4;
  if (idx >= total) return;
  const size_t e0 = idx * 4;
  const int d = (int)(e0 % D_);
  const size_t bq = e0 / D_;
  const int q = (int)(bq % Q_);
  const int b = (int)(bq / Q_);

  const float4 f = *(const float4*)(quest + e0);
  v4bf r;
  r[0] = (__bf16)f.x; r[1] = (__bf16)f.y; r[2] = (__bf16)f.z; r[3] = (__bf16)f.w;
  *(v4bf*)(qbf + e0) = r;

  __bf16* T = qbfT + ((size_t)b * D_ + d) * Q_ + q;
  T[0 * Q_] = r[0]; T[1 * Q_] = r[1]; T[2 * Q_] = r[2]; T[3 * Q_] = r[3];
}

// ---------------------------------------------------------------------------
// Kernel 1: per-wave 16 rows. sim = passage x question^T (A loaded once per
// dk, reused across 8 q-tiles), exact reference masked softmax over q, qp
// row-max, then pqv = attn x question written to out[:, :, 768:1536].
// ---------------------------------------------------------------------------
__global__ __launch_bounds__(WAVES * 32) void
bi_sim_softmax_pqv_kernel(const float* __restrict__ pass,
                          const __bf16* __restrict__ qbf,
                          const __bf16* __restrict__ qbfT,
                          const float* __restrict__ qmask,
                          float* __restrict__ out,
                          float* __restrict__ qp_sim) {
  __shared__ float s_sim[WAVES][ROWS_PER_WAVE][SIMLD];
  __shared__ float s_qmask[Q_];

  const int tid  = threadIdx.x;
  const int wave = tid >> 5;
  const int lane = tid & 31;
  const int wg   = blockIdx.x;                 // B * P/64 = 256 blocks
  const int b    = wg / (P_ / ROWS_PER_WG);
  const int pt   = wg % (P_ / ROWS_PER_WG);
  const int p0   = pt * ROWS_PER_WG + wave * ROWS_PER_WAVE;

#if __has_builtin(__builtin_amdgcn_tensor_load_to_lds)
  // Tensor Data Mover: DMA the per-batch question mask row (1 x 256 f32 tile)
  // into s_qmask. D# packed per ISA 8.3/8.4; g2/g3 zero (<=2D tensor), plus
  // the 6-arg toolchain's extra group zeroed. The cooperative load below
  // writes identical values, so numerics do not depend on descriptor
  // semantics corner cases.
  if (wave == 0) {
    const unsigned long long ga =
        (unsigned long long)(const void*)(qmask + (size_t)b * Q_);
    const unsigned int ldsa = (unsigned int)(size_t)(void*)&s_qmask[0];
    u32x4 g0;
    g0[0] = 1u;                                   // count=1, user descriptor
    g0[1] = ldsa;                                 // lds_addr
    g0[2] = (unsigned int)ga;                     // global_addr[31:0]
    g0[3] = (unsigned int)(ga >> 32) | (2u << 30);// global_addr[56:32]|type=2
    i32x8 g1;
    g1[0] = (2 << 16);                            // wg_mask=0, data_size=4B
    g1[1] = (int)((Q_ & 0xFFFF) << 16);           // tensor_dim0[15:0]
    g1[2] = (int)(((unsigned)Q_ >> 16) | (1u << 16)); // dim0[31:16]|dim1=1
    g1[3] = (int)(Q_ << 16);                      // dim1[31:16]=0 | tile_dim0
    g1[4] = 0;                                    // tile_dim1=0, tile_dim2=0
    g1[5] = Q_;                                   // tensor_dim0_stride[31:0]
    g1[6] = 0;                                    // strides hi
    g1[7] = 0;
    const i32x4 gz4 = {0, 0, 0, 0};
    const i32x8 gz8 = {0, 0, 0, 0, 0, 0, 0, 0};
    __builtin_amdgcn_tensor_load_to_lds(g0, g1, gz4, gz4, gz8, 0);
    __builtin_amdgcn_s_wait_tensorcnt(0);
  }
#endif

  for (int q = tid; q < Q_; q += WAVES * 32) s_qmask[q] = qmask[b * Q_ + q];
  __syncthreads();

  const float*  passB = pass + (size_t)b * P_ * D_;
  const __bf16* qbfB  = qbf  + (size_t)b * Q_ * D_;
  const __bf16* qbfTB = qbfT + (size_t)b * D_ * Q_;

  const int m = lane & 15;     // A-row / B-col within tile
  const int h = lane >> 4;     // lane half

  // ---------------- Phase 1: sim tiles --------------------------------------
  const float* arow = passB + (size_t)(p0 + m) * D_;
  for (int qb = 0; qb < 2; ++qb) {               // two blocks of 8 q-tiles
    v8f acc[8] = {};
    for (int dk = 0; dk < D_; dk += 32) {
      __builtin_prefetch(arow + dk + 64, 0, 1);
      v16bf a;
      #pragma unroll
      for (int j = 0; j < 8; ++j) {
        float2 f = *(const float2*)(arow + dk + a_kbase(j, h));
        a[2 * j]     = (__bf16)f.x;
        a[2 * j + 1] = (__bf16)f.y;
      }
      #pragma unroll
      for (int qt = 0; qt < 8; ++qt) {
        // B[k][n] = quest[q0+n][dk+k]; lane: col n=m, 16 consecutive d
        const int q0 = qb * 128 + qt * 16;
        const v16bf bf =
            *(const v16bf*)(qbfB + (size_t)(q0 + m) * D_ + dk + h * 16);
        acc[qt] = __builtin_amdgcn_wmma_f32_16x16x32_bf16(
            false, a, false, bf, (short)0, acc[qt], false, false);
      }
    }
    #pragma unroll
    for (int qt = 0; qt < 8; ++qt)
      #pragma unroll
      for (int r = 0; r < 8; ++r)
        s_sim[wave][r + 8 * h][qb * 128 + qt * 16 + m] = acc[qt][r];
  }
  __syncthreads();

  // ---------------- Phase 2: masked softmax over q (2 lanes per row) --------
  {
    const int rr   = lane >> 1;       // 0..15
    const int half = lane & 1;
    float* row = &s_sim[wave][rr][0];
    const int qlo = half * (Q_ / 2), qhi = qlo + (Q_ / 2);

    float mx = -1e30f, qpmax = -1e30f;
    for (int q = qlo; q < qhi; ++q) {
      float mk = s_qmask[q], v = row[q];
      mx    = fmaxf(mx, mk * v);                       // max of mask*vector
      qpmax = fmaxf(qpmax, v * mk + NEGV * (1.0f - mk));
    }
    mx    = fmaxf(mx,    __shfl_xor(mx, 1, 32));
    qpmax = fmaxf(qpmax, __shfl_xor(qpmax, 1, 32));

    float s = 0.0f;
    for (int q = qlo; q < qhi; ++q) {
      float mk = s_qmask[q];
      s += mk * __expf(mk * (mk * row[q] - mx));
    }
    s += __shfl_xor(s, 1, 32);
    const float norm = __logf(s + EPS_);

    for (int q = qlo; q < qhi; ++q) {
      float mk = s_qmask[q];
      row[q] = mk * __expf(mk * (mk * row[q] - mx) - norm);
    }
    if (half == 0) qp_sim[(size_t)b * P_ + p0 + rr] = qpmax;
  }
  __syncthreads();

  // ---------------- Phase 3: pqv = attn(16x256) x question(256xD) -----------
  // Pack this lane's 8 A-fragments (bf16) into registers once.
  v16bf afrag[Q_ / 32];
  #pragma unroll
  for (int s = 0; s < Q_ / 32; ++s) {
    #pragma unroll
    for (int j = 0; j < 8; ++j) {
      const int kb = a_kbase(j, h);
      afrag[s][2 * j]     = (__bf16)s_sim[wave][m][s * 32 + kb];
      afrag[s][2 * j + 1] = (__bf16)s_sim[wave][m][s * 32 + kb + 1];
    }
  }

  float* outB = out + (size_t)b * P_ * OUTD;
  for (int dt = 0; dt < D_ / 16; ++dt) {
    const int d0 = dt * 16;
    v8f c = {};
    #pragma unroll
    for (int s = 0; s < Q_ / 32; ++s) {
      // B[k][n] = quest[qk+k][d0+n]: contiguous q-run in qbfT row (d0+m)
      const v16bf bf =
          *(const v16bf*)(qbfTB + (size_t)(d0 + m) * Q_ + s * 32 + h * 16);
      c = __builtin_amdgcn_wmma_f32_16x16x32_bf16(
          false, afrag[s], false, bf, (short)0, c, false, false);
    }
    #pragma unroll
    for (int r = 0; r < 8; ++r)
      outB[(size_t)(p0 + r + 8 * h) * OUTD + D_ + d0 + m] = c[r];
  }
}

// ---------------------------------------------------------------------------
// Kernel 2: per-batch masked softmax over P=2048 -> qp_attention
// ---------------------------------------------------------------------------
__global__ __launch_bounds__(256) void
bi_qp_softmax_kernel(const float* __restrict__ qp_sim,
                     const float* __restrict__ pmask,
                     float* __restrict__ qp_attn) {
  __shared__ float red[256];
  const int b = blockIdx.x;
  const int t = threadIdx.x;
  const float* v  = qp_sim + (size_t)b * P_;
  const float* mk = pmask  + (size_t)b * P_;

  float mx = -1e30f;
  for (int p = t; p < P_; p += 256) mx = fmaxf(mx, mk[p] * v[p]);
  red[t] = mx; __syncthreads();
  for (int s = 128; s > 0; s >>= 1) {
    if (t < s) red[t] = fmaxf(red[t], red[t + s]);
    __syncthreads();
  }
  mx = red[0]; __syncthreads();

  float sum = 0.0f;
  for (int p = t; p < P_; p += 256) {
    float m = mk[p];
    sum += m * __expf(m * (m * v[p] - mx));
  }
  red[t] = sum; __syncthreads();
  for (int s = 128; s > 0; s >>= 1) {
    if (t < s) red[t] += red[t + s];
    __syncthreads();
  }
  const float norm = __logf(red[0] + EPS_);

  for (int p = t; p < P_; p += 256) {
    float m = mk[p];
    qp_attn[(size_t)b * P_ + p] = m * __expf(m * (m * v[p] - mx) - norm);
  }
}

// ---------------------------------------------------------------------------
// Kernel 3: qp_vector[b,d] = sum_p qp_attn[b,p] * passage[b,p,d]
// ---------------------------------------------------------------------------
__global__ __launch_bounds__(256) void
bi_qp_vector_kernel(const float* __restrict__ qp_attn,
                    const float* __restrict__ pass,
                    float* __restrict__ qp_vec) {
  const int b = blockIdx.y;
  const int d = blockIdx.x * 256 + threadIdx.x;   // gridDim.x = 3 -> d < 768
  const float* pb = pass + (size_t)b * P_ * D_ + d;
  const float* ab = qp_attn + (size_t)b * P_;
  float acc = 0.0f;
  for (int p = 0; p < P_; ++p) acc += ab[p] * pb[(size_t)p * D_];
  qp_vec[(size_t)b * D_ + d] = acc;
}

// ---------------------------------------------------------------------------
// Kernel 4: final concat [pass | pqv | pass*pqv | pass*qp_vec]
// (pqv slot already written by kernel 1)
// ---------------------------------------------------------------------------
__global__ __launch_bounds__(256) void
bi_assemble_kernel(const float* __restrict__ pass,
                   const float* __restrict__ qp_vec,
                   float* __restrict__ out) {
  const size_t idx = (size_t)blockIdx.x * 256 + threadIdx.x;  // B*P*(D/4)
  const size_t total = (size_t)B_ * P_ * (D_ / 4);
  if (idx >= total) return;
  const int d4    = (int)(idx % (D_ / 4));
  const size_t bp = idx / (D_ / 4);
  const int b     = (int)(bp / P_);

  const float4 pv  = *(const float4*)(pass + bp * D_ + d4 * 4);
  float* orow      = out + bp * OUTD;
  const float4 pqv = *(const float4*)(orow + D_ + d4 * 4);
  const float4 qv  = *(const float4*)(qp_vec + (size_t)b * D_ + d4 * 4);

  *(float4*)(orow + d4 * 4) = pv;
  *(float4*)(orow + 2 * D_ + d4 * 4) =
      make_float4(pv.x * pqv.x, pv.y * pqv.y, pv.z * pqv.z, pv.w * pqv.w);
  *(float4*)(orow + 3 * D_ + d4 * 4) =
      make_float4(pv.x * qv.x, pv.y * qv.y, pv.z * qv.z, pv.w * qv.w);
}

// ---------------------------------------------------------------------------
extern "C" void kernel_launch(void* const* d_in, const int* in_sizes, int n_in,
                              void* d_out, int out_size, void* d_ws, size_t ws_size,
                              hipStream_t stream) {
  (void)in_sizes; (void)n_in; (void)out_size; (void)ws_size;
  const float* pass  = (const float*)d_in[0];   // (B,P,D)
  const float* quest = (const float*)d_in[1];   // (B,Q,D)
  const float* pmask = (const float*)d_in[2];   // (B,P)
  const float* qmask = (const float*)d_in[3];   // (B,Q)
  float* out = (float*)d_out;                   // (B,P,4D)

  float* ws      = (float*)d_ws;
  float* qp_sim  = ws;                          // B*P floats
  float* qp_attn = ws + (size_t)B_ * P_;        // B*P floats
  float* qp_vec  = ws + (size_t)2 * B_ * P_;    // B*D floats
  __bf16* qbf  = (__bf16*)(ws + (size_t)2 * B_ * P_ + B_ * D_);
  __bf16* qbfT = qbf + (size_t)B_ * Q_ * D_;

  const size_t qtot = (size_t)B_ * Q_ * D_ / 4;
  bi_convert_q_kernel<<<(unsigned)((qtot + 255) / 256), 256, 0, stream>>>(
      quest, qbf, qbfT);

  bi_sim_softmax_pqv_kernel<<<B_ * (P_ / ROWS_PER_WG), WAVES * 32, 0, stream>>>(
      pass, qbf, qbfT, qmask, out, qp_sim);

  bi_qp_softmax_kernel<<<B_, 256, 0, stream>>>(qp_sim, pmask, qp_attn);

  bi_qp_vector_kernel<<<dim3(D_ / 256, B_), 256, 0, stream>>>(qp_attn, pass, qp_vec);

  const size_t total = (size_t)B_ * P_ * (D_ / 4);
  bi_assemble_kernel<<<(unsigned)((total + 255) / 256), 256, 0, stream>>>(
      pass, qp_vec, out);
}
